// MPNN_HeteroGNN_90220083020174
// MI455X (gfx1250) — matrix operand order repaired and checked by reading, not compiled, not run on hardware
//
#include <hip/hip_runtime.h>
#include <hip/hip_bf16.h>

// ---------------------------------------------------------------------------
// MPNN hetero-GNN forward for MI455X (gfx1250, wave32, WMMA).
//
//  * One WMMA gather-GEMM kernel for every dense linear:
//      - A rows gathered by index, f32->f16 converted into LDS
//        (32 x Kpad tile, +16B row pad -> conflict-free ds_load_b128)
//      - B = pre-transposed f16 weights Wt[n][k] (L2-resident)
//      - 8 waves/block -> 32x128 C tile; each wave holds TWO fp32 accumulators
//        and issues two v_wmma_f32_16x16x32_f16 per B fragment (2x arithmetic
//        intensity on the L0/L2 weight path vs a 16-row tile)
//      - epilogue: bias + {none|relu|softplus}, optional fused scatter via
//        global_atomic_add_f32 (fuses segment_sum into the GEMM)
//  * Attention Q/K and node-dependent parts of V1 / edge-MLP l1 hoisted to
//    node-level GEMMs (q depends only on src node, etc.) -> ~2x fewer FLOPs
//    than the reference's per-edge projections.
//  * Segment softmax: order-preserving float->uint atomicMax + atomicAdd.
//
// Param layout assumption: d_in[0..12] = the 13 data arrays in setup_inputs()
// order; d_in[13..] = jax pytree leaves of `params` (dict keys sorted
// alphabetically, lists in order; {"w","b"} -> b,w ; {"g","b"} -> b,g).
// ---------------------------------------------------------------------------

typedef __attribute__((ext_vector_type(16))) _Float16 v16h;
typedef __attribute__((ext_vector_type(8)))  _Float16 v8h;
typedef __attribute__((ext_vector_type(8)))  float    v8f;

#define H 128
#define HEADS 4
#define N_RES 20000
#define N_LIG 10000
#define E_PEP 40000
#define E_PP 200000
#define E_PPGEO 200000
#define E_LIG 20000
#define E_PL 100000
#define E_PLGEO 100000
#define T_PP 8
#define T_PL 8
#define D_PPGEO 16
#define D_PLGEO 5
#define RES_IN 21
#define LIG_IN 10
#define NUM_BLOCKS 3

#define KPAD_MAX 256
#define LDSROW (KPAD_MAX + 8)
#define MT 32   // M tile per block (two 16-row WMMA tiles)

// ---------------- device helpers ----------------
__device__ __forceinline__ unsigned fkey(float f) {
    unsigned u = __float_as_uint(f);
    return (u & 0x80000000u) ? ~u : (u | 0x80000000u);
}
__device__ __forceinline__ float funkey(unsigned u) {
    return __uint_as_float((u & 0x80000000u) ? (u & 0x7fffffffu) : ~u);
}
__device__ __forceinline__ float lrelu02(float x) { return x > 0.f ? x : 0.2f * x; }

// ---------------- weight convert: W[K,N] f32 -> Wt[Npad][Kpad] f16 ----------
__global__ void k_cvt_wt(const float* __restrict__ W, int K0r, int K, int N,
                         _Float16* __restrict__ Wt, int Kpad, int Npad) {
    long i = (long)blockIdx.x * 256 + threadIdx.x;
    long tot = (long)Kpad * Npad;
    if (i >= tot) return;
    int n = (int)(i / Kpad), k = (int)(i - (long)n * Kpad);
    float v = (n < N && k < K) ? W[(long)(K0r + k) * N + n] : 0.f;
    Wt[i] = (_Float16)v;
}

// ---------------- the WMMA gather-GEMM (32x128 tile / block) ----------------
__global__ __launch_bounds__(256) void k_gemm(
    const float* __restrict__ A0, const int* __restrict__ i0, int K0,
    const float* __restrict__ A1, const int* __restrict__ i1, int K1,
    const float* __restrict__ A2, const int* __restrict__ i2, int K2,
    const _Float16* __restrict__ Wt, int Kpad,
    const float* __restrict__ bias,
    float* __restrict__ C, int ldC, const int* __restrict__ scat,
    int M, int N, int Npad, int act) {
    __shared__ __align__(16) _Float16 As[MT * LDSROW];
    __shared__ int rows[3][MT];
    const int tid = threadIdx.y * 32 + threadIdx.x;
    const int m0 = blockIdx.x * MT;
    const int RS = Kpad + 8;
    if (tid < 3 * MT) {
        int sg = tid / MT, r = tid - sg * MT, row = m0 + r, g = -1;
        const int* ix = (sg == 0) ? i0 : (sg == 1) ? i1 : i2;
        int Ks = (sg == 0) ? K0 : (sg == 1) ? K1 : K2;
        if (Ks > 0 && row < M) g = ix ? ix[row] : row;
        rows[sg][r] = g;
    }
    __syncthreads();
    const int K01 = K0 + K1, Kt = K01 + K2;
    for (int e = tid; e < MT * Kpad; e += 256) {
        int r = e / Kpad, k = e - r * Kpad;
        float v = 0.f;
        if (k < K0)       { int g = rows[0][r]; if (g >= 0) v = A0[(long)g * K0 + k]; }
        else if (k < K01) { int g = rows[1][r]; if (g >= 0) v = A1[(long)g * K1 + (k - K0)]; }
        else if (k < Kt)  { int g = rows[2][r]; if (g >= 0) v = A2[(long)g * K2 + (k - K01)]; }
        As[r * RS + k] = (_Float16)v;
    }
    __syncthreads();
    const int lane = threadIdx.x;
    const int n0 = blockIdx.y * 128 + threadIdx.y * 16;
    if (n0 >= Npad) return;                 // wave-uniform: safe vs WMMA EXEC rule
    const int nl = n0 + (lane & 15);
    const int hi = (lane < 16) ? 0 : 1;
    const _Float16* arow0 = &As[(lane & 15) * RS + hi * 8];
    const _Float16* arow1 = arow0 + 16 * RS;
    const _Float16* brow = Wt + (long)nl * Kpad + hi * 8;
    v8f acc0 = {}, acc1 = {};
    for (int ks = 0; ks < Kpad; ks += 32) {
        union { v16h v; v8h h[2]; } a0, a1, b;
        b.h[0]  = *(const v8h*)(brow + ks);        // K chunk 0..7 (lane phase)
        b.h[1]  = *(const v8h*)(brow + ks + 16);
        a0.h[0] = *(const v8h*)(arow0 + ks);       // K chunk 0..7 (lane phase)
        a0.h[1] = *(const v8h*)(arow0 + ks + 16);  // K chunk 16..23
        a1.h[0] = *(const v8h*)(arow1 + ks);
        a1.h[1] = *(const v8h*)(arow1 + ks + 16);
        acc0 = __builtin_amdgcn_wmma_f32_16x16x32_f16(
            false, a0.v, false, b.v, (short)0, acc0, false, false);
        acc1 = __builtin_amdgcn_wmma_f32_16x16x32_f16(
            false, a1.v, false, b.v, (short)0, acc1, false, false);
    }
    if (nl >= N) return;
    float bv = bias ? bias[nl] : 0.f;
#pragma unroll
    for (int r = 0; r < 8; ++r) {
        int row = m0 + r + hi * 8;
        if (row < M) {
            float v = acc0[r] + bv;
            if (act == 1) v = fmaxf(v, 0.f);
            else if (act == 2) v = (v > 20.f) ? v : log1pf(__expf(v));
            if (scat) atomicAdd(&C[(long)scat[row] * ldC + nl], v);
            else C[(long)row * ldC + nl] = v;
        }
    }
#pragma unroll
    for (int r = 0; r < 8; ++r) {
        int row = m0 + 16 + r + hi * 8;
        if (row < M) {
            float v = acc1[r] + bv;
            if (act == 1) v = fmaxf(v, 0.f);
            else if (act == 2) v = (v > 20.f) ? v : log1pf(__expf(v));
            if (scat) atomicAdd(&C[(long)scat[row] * ldC + nl], v);
            else C[(long)row * ldC + nl] = v;
        }
    }
}

// ---------------- elementwise / reduction helpers ----------------
__global__ void k_fillf(float* p, long n, float v) {
    long i = (long)blockIdx.x * 256 + threadIdx.x;
    if (i < n) p[i] = v;
}
__global__ void k_fillu(unsigned* p, long n, unsigned v) {
    long i = (long)blockIdx.x * 256 + threadIdx.x;
    if (i < n) p[i] = v;
}

__global__ void k_rowdot(const float* __restrict__ x, const float* __restrict__ vec,
                         float* __restrict__ out) {
    __shared__ float sb[128];
    int r = blockIdx.x, t = threadIdx.x;
    sb[t] = x[(long)r * H + t] * vec[t];
    __syncthreads();
    for (int o = 64; o; o >>= 1) { if (t < o) sb[t] += sb[t + o]; __syncthreads(); }
    if (t == 0) out[r] = sb[0];
}

// GAT passes (edges + self loops appended)
__global__ void k_gat_max(const float* as, const float* ad, const int* src,
                          const int* dst, int E, int nN, unsigned* m) {
    int i = blockIdx.x * 256 + threadIdx.x;
    if (i >= E + nN) return;
    int s, d;
    if (i < E) { s = src[i]; d = dst[i]; } else s = d = i - E;
    atomicMax(&m[d], fkey(lrelu02(as[s] + ad[d])));
}
__global__ void k_gat_sum(const float* as, const float* ad, const int* src,
                          const int* dst, int E, int nN, const unsigned* m, float* ss) {
    int i = blockIdx.x * 256 + threadIdx.x;
    if (i >= E + nN) return;
    int s, d;
    if (i < E) { s = src[i]; d = dst[i]; } else s = d = i - E;
    atomicAdd(&ss[d], __expf(lrelu02(as[s] + ad[d]) - funkey(m[d])));
}
__global__ void k_gat_scat(const float* __restrict__ xl, const float* as, const float* ad,
                           const int* src, const int* dst, int E, int nN,
                           const unsigned* m, const float* ss, float* macc) {
    int i = blockIdx.x, t = threadIdx.x;
    int s, d;
    if (i < E) { s = src[i]; d = dst[i]; } else s = d = i - E;
    float w = __expf(lrelu02(as[s] + ad[d]) - funkey(m[d])) / ss[d];
    atomicAdd(&macc[(long)d * H + t], w * xl[(long)s * H + t]);
}

// per-edge multi-head q.k logits
__global__ void k_amp_logits(const float* __restrict__ Q, const float* __restrict__ K,
                             const int* src, const int* dst, const float* scale,
                             int heads, float* la) {
    __shared__ float sb[128];
    int e = blockIdx.x, t = threadIdx.x;
    int s = src[e], d = dst[e];
    sb[t] = Q[(long)s * H + t] * K[(long)d * H + t];
    __syncthreads();
    int dh = H / heads;
    for (int o = dh >> 1; o; o >>= 1) {
        if ((t & (dh - 1)) < o) sb[t] += sb[t + o];
        __syncthreads();
    }
    if ((t & (dh - 1)) == 0) la[(long)e * heads + t / dh] = sb[t] * scale[t / dh];
}
__global__ void k_amp_max(const float* la, const int* dst, int heads, unsigned* m, long n) {
    long i = (long)blockIdx.x * 256 + threadIdx.x;
    if (i >= n) return;
    long e = i / heads; int h = (int)(i - e * heads);
    atomicMax(&m[(long)dst[e] * heads + h], fkey(la[i]));
}
__global__ void k_amp_sum(const float* la, const int* dst, int heads,
                          const unsigned* m, float* ss, long n) {
    long i = (long)blockIdx.x * 256 + threadIdx.x;
    if (i >= n) return;
    long e = i / heads; int h = (int)(i - e * heads);
    long ih = (long)dst[e] * heads + h;
    atomicAdd(&ss[ih], __expf(la[i] - funkey(m[ih])));
}
__global__ void k_scalev(const float* __restrict__ v, const float* la, const unsigned* m,
                         const float* ss, const int* dst, int heads, float* vw) {
    int e = blockIdx.x, t = threadIdx.x;
    int dh = H / heads, h = t / dh, d = dst[e];
    long ih = (long)d * heads + h;
    float w = __expf(la[(long)e * heads + h] - funkey(m[ih])) / ss[ih];
    vw[(long)e * H + t] = v[(long)e * H + t] * w;
}

// y[r] = relu(a[ia[r]] + b[ib[r]] + c[r] + bias), width = blockDim.x
__global__ void k_gadd(const float* __restrict__ a, const int* __restrict__ ia,
                       const float* __restrict__ b, const int* __restrict__ ib,
                       const float* __restrict__ c, const float* __restrict__ bias,
                       float* __restrict__ y) {
    int r = blockIdx.x, t = threadIdx.x, Hw = blockDim.x;
    float v = a[(long)ia[r] * Hw + t] + b[(long)ib[r] * Hw + t];
    if (c) v += c[(long)r * Hw + t];
    if (bias) v += bias[t];
    y[(long)r * Hw + t] = fmaxf(v, 0.f);
}

// y = act(LN(x + add + vec)) width = blockDim.x (128/256)
__global__ void k_ln(const float* __restrict__ x, const float* __restrict__ add,
                     const float* __restrict__ vec, const float* g, const float* b,
                     int relu, float* __restrict__ y) {
    __shared__ float sb[256];
    int r = blockIdx.x, t = threadIdx.x, Hw = blockDim.x;
    float v = x[(long)r * Hw + t];
    if (add) v += add[(long)r * Hw + t];
    if (vec) v += vec[t];
    sb[t] = v; __syncthreads();
    for (int o = Hw >> 1; o; o >>= 1) { if (t < o) sb[t] += sb[t + o]; __syncthreads(); }
    float mu = sb[0] / Hw; __syncthreads();
    float dv = v - mu;
    sb[t] = dv * dv; __syncthreads();
    for (int o = Hw >> 1; o; o >>= 1) { if (t < o) sb[t] += sb[t + o]; __syncthreads(); }
    float var = sb[0] / Hw;
    float o_ = dv * rsqrtf(var + 1e-5f) * g[t] + b[t];
    if (relu) o_ = fmaxf(o_, 0.f);
    y[(long)r * Hw + t] = o_;
}

// ---------------- host-side param bookkeeping ----------------
struct PLin { const float* w; const float* b; };
struct PLN  { const float* g; const float* b; };
struct PAmp { PLin edge_proj, k, out, q, v1, v2; const float* scale; };
struct PGat { const float* att_dst; const float* att_src; const float* bias; const float* lin_w; };
struct PMlp { PLin l1, l2; };
struct PBlk {
    PAmp cross, cross_geo; PLN cross_norm; PGat lig_gat; PLN lig_norm;
    PMlp pl_edge, pp_edge; PGat res_gat; PAmp res_geo, res_int; PLN res_norm;
};
struct AmpW { _Float16 *ep, *q, *k, *v1a, *v1b, *v1c, *v2, *out; int Ed; };
struct EdgeMlpW { _Float16 *a, *b, *c, *l2; };   // 2H+T -> 128 -> T
struct HeadW    { _Float16 *a, *b, *c, *l2; };   // 2H+T -> 256 -> T

extern "C" void kernel_launch(void* const* d_in, const int* in_sizes, int n_in,
                              void* d_out, int out_size, void* d_ws, size_t ws_size,
                              hipStream_t stream) {
    (void)in_sizes; (void)n_in; (void)out_size; (void)ws_size;
    // ---- data arrays ----
    const float* x_residue = (const float*)d_in[0];
    const float* x_ligand  = (const float*)d_in[1];
    const int* ei_pep   = (const int*)d_in[2];
    const int* ei_pp    = (const int*)d_in[3];
    const int* ei_ppgeo = (const int*)d_in[4];
    const int* ei_lig   = (const int*)d_in[5];
    const int* ei_pl    = (const int*)d_in[6];
    const int* ei_plgeo = (const int*)d_in[7];
    const float* ea_pp    = (const float*)d_in[8];
    const float* ea_ppgeo = (const float*)d_in[9];
    const float* ea_pl    = (const float*)d_in[10];
    const float* ea_plgeo = (const float*)d_in[11];
    const int* pep_s = ei_pep, *pep_d = ei_pep + E_PEP;
    const int* pp_s = ei_pp, *pp_d = ei_pp + E_PP;
    const int* geo_s = ei_ppgeo, *geo_d = ei_ppgeo + E_PPGEO;
    const int* lg_s = ei_lig, *lg_d = ei_lig + E_LIG;
    const int* pl_s = ei_pl, *pl_d = ei_pl + E_PL;
    const int* plg_s = ei_plgeo, *plg_d = ei_plgeo + E_PLGEO;

    // ---- param leaves (pytree order) ----
    int pi = 13;
    auto F   = [&]() { return (const float*)d_in[pi++]; };
    auto rLin = [&](PLin& L) { L.b = F(); L.w = F(); };
    auto rLN  = [&](PLN& L)  { L.b = F(); L.g = F(); };
    auto rAmp = [&](PAmp& a) { rLin(a.edge_proj); rLin(a.k); rLin(a.out); rLin(a.q);
                               a.scale = F(); rLin(a.v1); rLin(a.v2); };
    auto rGat = [&](PGat& g) { g.att_dst = F(); g.att_src = F(); g.bias = F(); g.lin_w = F(); };
    auto rMlp = [&](PMlp& m) { rLin(m.l1); rLin(m.l2); };
    PBlk blk[NUM_BLOCKS];
    for (int i = 0; i < NUM_BLOCKS; ++i) {
        PBlk& b = blk[i];
        rAmp(b.cross); rAmp(b.cross_geo); rLN(b.cross_norm); rGat(b.lig_gat); rLN(b.lig_norm);
        rMlp(b.pl_edge); rMlp(b.pp_edge); rGat(b.res_gat); rAmp(b.res_geo); rAmp(b.res_int);
        rLN(b.res_norm);
    }
    PLin cls1, cls2, cls_final; PLN cls_ln1, cls_ln2; PLin lig_proj; PLN lig_proj_ln;
    PMlp pl_count, pl_zero, pp_count, pp_zero; PLin res_proj; PLN res_proj_ln;
    rLin(cls1); rLin(cls2); rLin(cls_final); rLN(cls_ln1); rLN(cls_ln2);
    rLin(lig_proj); rLN(lig_proj_ln);
    rMlp(pl_count); rMlp(pl_zero); rMlp(pp_count); rMlp(pp_zero);
    rLin(res_proj); rLN(res_proj_ln);

    // ---- workspace bump allocator ----
    char* wsp = (char*)d_ws;
    auto alloc = [&](size_t bytes) -> void* {
        void* p = wsp; wsp += (bytes + 255) & ~(size_t)255; return p;
    };
    auto cvt = [&](const float* W, int K0r, int K, int N, int Kpad, int Npad) -> _Float16* {
        _Float16* dst = (_Float16*)alloc((size_t)Kpad * Npad * sizeof(_Float16));
        long tot = (long)Kpad * Npad;
        k_cvt_wt<<<(unsigned)((tot + 255) / 256), 256, 0, stream>>>(W, K0r, K, N, dst, Kpad, Npad);
        return dst;
    };

    // ---- convert all weights to f16, transposed/padded ----
    AmpW aw_geo[3], aw_int[3], aw_c[3], aw_cg[3];
    _Float16 *gw_res[3], *gw_lig[3];
    EdgeMlpW ew_pp[3], ew_pl[3];
    auto cvtAmp = [&](const PAmp& a, int Ed) {
        AmpW w; w.Ed = Ed;
        w.ep  = cvt(a.edge_proj.w, 0, Ed, H, 32, H);
        w.q   = cvt(a.q.w, 0, H, H, H, H);
        w.k   = cvt(a.k.w, 0, H, H, H, H);
        w.v1a = cvt(a.v1.w, 0, H, H, H, H);
        w.v1b = cvt(a.v1.w, H, H, H, H, H);
        w.v1c = cvt(a.v1.w, 2 * H, H, H, H, H);
        w.v2  = cvt(a.v2.w, 0, H, H, H, H);
        w.out = cvt(a.out.w, 0, H, H, H, H);
        return w;
    };
    auto cvtEMlp = [&](const PMlp& m, int T) {
        EdgeMlpW w;
        w.a  = cvt(m.l1.w, 0, H, H, H, H);
        w.b  = cvt(m.l1.w, H, H, H, H, H);
        w.c  = cvt(m.l1.w, 2 * H, T, H, 32, H);
        w.l2 = cvt(m.l2.w, 0, H, T, H, 16);
        return w;
    };
    for (int i = 0; i < NUM_BLOCKS; ++i) {
        aw_geo[i] = cvtAmp(blk[i].res_geo, D_PPGEO);
        aw_int[i] = cvtAmp(blk[i].res_int, T_PP);
        aw_c[i]   = cvtAmp(blk[i].cross, T_PL);
        aw_cg[i]  = cvtAmp(blk[i].cross_geo, D_PLGEO);
        gw_res[i] = cvt(blk[i].res_gat.lin_w, 0, H, H, H, H);
        gw_lig[i] = cvt(blk[i].lig_gat.lin_w, 0, H, H, H, H);
        ew_pp[i]  = cvtEMlp(blk[i].pp_edge, T_PP);
        ew_pl[i]  = cvtEMlp(blk[i].pl_edge, T_PL);
    }
    _Float16* wt_resproj = cvt(res_proj.w, 0, RES_IN, H, 32, H);
    _Float16* wt_ligproj = cvt(lig_proj.w, 0, LIG_IN, H, 32, H);
    _Float16* wt_cls1 = cvt(cls1.w, 0, H, 256, H, 256);
    _Float16* wt_cls2 = cvt(cls2.w, 0, 256, H, 256, H);
    _Float16* wt_clsf = cvt(cls_final.w, 0, H, 20, H, 32);
    auto cvtHead = [&](const PMlp& m, int T) {
        HeadW w;
        w.a  = cvt(m.l1.w, 0, H, 256, H, 256);
        w.b  = cvt(m.l1.w, H, H, 256, H, 256);
        w.c  = cvt(m.l1.w, 2 * H, T, 256, 32, 256);
        w.l2 = cvt(m.l2.w, 0, 256, T, 256, 16);
        return w;
    };
    HeadW hw_ppc = cvtHead(pp_count, T_PP), hw_ppz = cvtHead(pp_zero, T_PP);
    HeadW hw_plc = cvtHead(pl_count, T_PL), hw_plz = cvtHead(pl_zero, T_PL);

    // ---- activation buffers ----
    auto af = [&](long n) { return (float*)alloc((size_t)n * 4); };
    float *xrA = af((long)N_RES * H), *xrB = af((long)N_RES * H);
    float *xlA = af((long)N_LIG * H), *xlB = af((long)N_LIG * H);
    float *macc = af((long)N_RES * H), *maccL = af((long)N_LIG * H);
    float *tn0 = af((long)N_RES * H), *tn1 = af((long)N_RES * H);
    float *tn2 = af((long)N_RES * H), *tn3 = af((long)N_RES * H);
    float *nbig0 = af((long)N_RES * 256), *nbig1 = af((long)N_RES * 256);
    float *EB0 = af((long)E_PP * H), *EB1 = af((long)E_PP * H);
    float *la = af((long)E_PP * HEADS);
    unsigned* segm = (unsigned*)alloc((size_t)N_RES * HEADS * 4);
    float* segs = af((long)N_RES * HEADS);
    float *sa = af(N_RES), *sdv = af(N_RES);
    float *ppA = af((long)E_PP * T_PP), *ppB = af((long)E_PP * T_PP);
    float *plA = af((long)E_PL * T_PL), *plB = af((long)E_PL * T_PL);
    float* out = (float*)d_out;

    auto fillf = [&](float* p, long n, float v) {
        k_fillf<<<(unsigned)((n + 255) / 256), 256, 0, stream>>>(p, n, v);
    };
    auto fillu = [&](unsigned* p, long n) {
        k_fillu<<<(unsigned)((n + 255) / 256), 256, 0, stream>>>(p, n, 0u);
    };
    auto gemm1 = [&](const float* A, const int* ia, int K, int Kpad, const _Float16* Wt,
                     const float* bias, float* C, int ldC, const int* scat,
                     int M, int N, int Npad, int act) {
        dim3 g((M + MT - 1) / MT, (Npad + 127) / 128), bk(32, 8);
        k_gemm<<<g, bk, 0, stream>>>(A, ia, K, nullptr, nullptr, 0, nullptr, nullptr, 0,
                                     Wt, Kpad, bias, C, ldC, scat, M, N, Npad, act);
    };

    // ---- attentive message passing (factored) ----
    auto runAmp = [&](const PAmp& a, const AmpW& w, const float* xs, int nSrc,
                      const float* xd, int nDst, const int* src, const int* dst, int E,
                      const float* ea, int heads, float* mac) {
        gemm1(xs, nullptr, H, H, w.q, a.q.b, tn0, H, nullptr, nSrc, H, H, 0);      // Qn
        gemm1(xd, nullptr, H, H, w.k, a.k.b, tn1, H, nullptr, nDst, H, H, 0);      // Kn
        gemm1(xd, nullptr, H, H, w.v1a, nullptr, tn2, H, nullptr, nDst, H, H, 0);  // W1a x_i
        gemm1(xs, nullptr, H, H, w.v1b, nullptr, tn3, H, nullptr, nSrc, H, H, 0);  // W1b x_j
        gemm1(ea, nullptr, w.Ed, 32, w.ep, a.edge_proj.b, EB0, H, nullptr, E, H, H, 0); // ep
        gemm1(EB0, nullptr, H, H, w.v1c, nullptr, EB1, H, nullptr, E, H, H, 0);    // W1c ep
        k_amp_logits<<<E, 128, 0, stream>>>(tn0, tn1, src, dst, a.scale, heads, la);
        fillu(segm, (long)nDst * heads);
        fillf(segs, (long)nDst * heads, 0.f);
        long n = (long)E * heads;
        k_amp_max<<<(unsigned)((n + 255) / 256), 256, 0, stream>>>(la, dst, heads, segm, n);
        k_amp_sum<<<(unsigned)((n + 255) / 256), 256, 0, stream>>>(la, dst, heads, segm, segs, n);
        k_gadd<<<E, 128, 0, stream>>>(tn2, dst, tn3, src, EB1, a.v1.b, EB0);       // v1 hidden
        gemm1(EB0, nullptr, H, H, w.v2, a.v2.b, EB1, H, nullptr, E, H, H, 0);      // v
        k_scalev<<<E, 128, 0, stream>>>(EB1, la, segm, segs, dst, heads, EB0);     // v*w
        gemm1(EB0, nullptr, H, H, w.out, a.out.b, mac, H, dst, E, H, H, 0);        // scatter
    };
    auto runGat = [&](const PGat& g, const _Float16* lw, const float* x, int nN,
                      const int* src, const int* dst, int E, float* mac) {
        gemm1(x, nullptr, H, H, lw, nullptr, tn0, H, nullptr, nN, H, H, 0);        // xl
        k_rowdot<<<nN, 128, 0, stream>>>(tn0, g.att_src, sa);
        k_rowdot<<<nN, 128, 0, stream>>>(tn0, g.att_dst, sdv);
        fillu(segm, nN);
        fillf(segs, nN, 0.f);
        int tot = E + nN;
        k_gat_max<<<(tot + 255) / 256, 256, 0, stream>>>(sa, sdv, src, dst, E, nN, segm);
        k_gat_sum<<<(tot + 255) / 256, 256, 0, stream>>>(sa, sdv, src, dst, E, nN, segm, segs);
        k_gat_scat<<<tot, 128, 0, stream>>>(tn0, sa, sdv, src, dst, E, nN, segm, segs, mac);
    };

    // ---- input projections ----
    gemm1(x_residue, nullptr, RES_IN, 32, wt_resproj, res_proj.b, tn0, H, nullptr, N_RES, H, H, 0);
    k_ln<<<N_RES, 128, 0, stream>>>(tn0, nullptr, nullptr, res_proj_ln.g, res_proj_ln.b, 0, xrA);
    gemm1(x_ligand, nullptr, LIG_IN, 32, wt_ligproj, lig_proj.b, tn0, H, nullptr, N_LIG, H, H, 0);
    k_ln<<<N_LIG, 128, 0, stream>>>(tn0, nullptr, nullptr, lig_proj_ln.g, lig_proj_ln.b, 0, xlA);

    hipMemcpyAsync(ppA, ea_pp, (size_t)E_PP * T_PP * 4, hipMemcpyDeviceToDevice, stream);
    hipMemcpyAsync(plA, ea_pl, (size_t)E_PL * T_PL * 4, hipMemcpyDeviceToDevice, stream);
    float *xr = xrA, *xr2 = xrB, *xl = xlA, *xl2 = xlB;
    float *cpp = ppA, *cppN = ppB, *cpl = plA, *cplN = plB;

    for (int bi = 0; bi < NUM_BLOCKS; ++bi) {
        const PBlk& b = blk[bi];
        // residue messages (all from old x_res)
        fillf(macc, (long)N_RES * H, 0.f);
        runGat(b.res_gat, gw_res[bi], xr, N_RES, pep_s, pep_d, E_PEP, macc);
        runAmp(b.res_geo, aw_geo[bi], xr, N_RES, xr, N_RES, geo_s, geo_d, E_PPGEO,
               ea_ppgeo, HEADS, macc);
        runAmp(b.res_int, aw_int[bi], xr, N_RES, xr, N_RES, pp_s, pp_d, E_PP,
               cpp, HEADS, macc);
        k_ln<<<N_RES, 128, 0, stream>>>(xr, macc, b.res_gat.bias, b.res_norm.g, b.res_norm.b, 0, xr2);
        { float* t = xr; xr = xr2; xr2 = t; }
        // ligand
        fillf(maccL, (long)N_LIG * H, 0.f);
        runGat(b.lig_gat, gw_lig[bi], xl, N_LIG, lg_s, lg_d, E_LIG, maccL);
        k_ln<<<N_LIG, 128, 0, stream>>>(xl, maccL, b.lig_gat.bias, b.lig_norm.g, b.lig_norm.b, 0, xl2);
        { float* t = xl; xl = xl2; xl2 = t; }
        // cross (lig -> res)
        fillf(macc, (long)N_RES * H, 0.f);
        runAmp(b.cross_geo, aw_cg[bi], xl, N_LIG, xr, N_RES, plg_s, plg_d, E_PLGEO,
               ea_plgeo, 1, macc);
        runAmp(b.cross, aw_c[bi], xl, N_LIG, xr, N_RES, pl_s, pl_d, E_PL, cpl, HEADS, macc);
        k_ln<<<N_RES, 128, 0, stream>>>(xr, macc, nullptr, b.cross_norm.g, b.cross_norm.b, 0, xr2);
        { float* t = xr; xr = xr2; xr2 = t; }
        // edge feature updates (use updated node feats)
        gemm1(xr, nullptr, H, H, ew_pp[bi].a, nullptr, tn0, H, nullptr, N_RES, H, H, 0);
        gemm1(xr, nullptr, H, H, ew_pp[bi].b, nullptr, tn1, H, nullptr, N_RES, H, H, 0);
        gemm1(cpp, nullptr, T_PP, 32, ew_pp[bi].c, nullptr, EB0, H, nullptr, E_PP, H, H, 0);
        k_gadd<<<E_PP, 128, 0, stream>>>(tn0, pp_s, tn1, pp_d, EB0, b.pp_edge.l1.b, EB1);
        gemm1(EB1, nullptr, H, H, ew_pp[bi].l2, b.pp_edge.l2.b, cppN, T_PP, nullptr,
              E_PP, T_PP, 16, 0);
        gemm1(xl, nullptr, H, H, ew_pl[bi].a, nullptr, tn0, H, nullptr, N_LIG, H, H, 0);
        gemm1(xr, nullptr, H, H, ew_pl[bi].b, nullptr, tn1, H, nullptr, N_RES, H, H, 0);
        gemm1(cpl, nullptr, T_PL, 32, ew_pl[bi].c, nullptr, EB0, H, nullptr, E_PL, H, H, 0);
        k_gadd<<<E_PL, 128, 0, stream>>>(tn0, pl_s, tn1, pl_d, EB0, b.pl_edge.l1.b, EB1);
        gemm1(EB1, nullptr, H, H, ew_pl[bi].l2, b.pl_edge.l2.b, cplN, T_PL, nullptr,
              E_PL, T_PL, 16, 0);
        { float* t = cpp; cpp = cppN; cppN = t; }
        { float* t = cpl; cpl = cplN; cplN = t; }
    }

    // ---- classifier heads ----
    const long OFF_FEAT = 400000, OFF_PPC = 2960000, OFF_PPZ = 4560000;
    const long OFF_PLC = 6160000, OFF_PLZ = 6960000;
    gemm1(xr, nullptr, H, H, wt_cls1, cls1.b, nbig0, 256, nullptr, N_RES, 256, 256, 0);
    k_ln<<<N_RES, 256, 0, stream>>>(nbig0, nullptr, nullptr, cls_ln1.g, cls_ln1.b, 1, nbig1);
    gemm1(nbig1, nullptr, 256, 256, wt_cls2, cls2.b, tn0, H, nullptr, N_RES, H, H, 0);
    k_ln<<<N_RES, 128, 0, stream>>>(tn0, nullptr, nullptr, cls_ln2.g, cls_ln2.b, 1, out + OFF_FEAT);
    gemm1(out + OFF_FEAT, nullptr, H, H, wt_clsf, cls_final.b, out, 20, nullptr, N_RES, 20, 32, 0);

    // ---- ZIP edge heads (chunked, 256-wide hidden) ----
    auto runHead = [&](const PMlp& m, const HeadW& w, const float* xa, const int* ia,
                       const float* xb, const int* ib, const float* cur, int T, int E,
                       float* outP, int act) {
        gemm1(xa, nullptr, H, H, w.a, nullptr, nbig0, 256, nullptr,
              (xa == xl) ? N_LIG : N_RES, 256, 256, 0);
        gemm1(xb, nullptr, H, H, w.b, nullptr, nbig1, 256, nullptr, N_RES, 256, 256, 0);
        const int CH = 100000;
        for (int c0 = 0; c0 < E; c0 += CH) {
            int cm = (E - c0 < CH) ? (E - c0) : CH;
            gemm1(cur + (long)c0 * T, nullptr, T, 32, w.c, nullptr, EB0, 256, nullptr,
                  cm, 256, 256, 0);
            k_gadd<<<cm, 256, 0, stream>>>(nbig0, ia + c0, nbig1, ib + c0, EB0, m.l1.b, EB1);
            gemm1(EB1, nullptr, 256, 256, w.l2, m.l2.b, outP + (long)c0 * T, T, nullptr,
                  cm, T, 16, act);
        }
    };
    runHead(pp_count, hw_ppc, xr, pp_s, xr, pp_d, cpp, T_PP, E_PP, out + OFF_PPC, 2);
    runHead(pp_zero,  hw_ppz, xr, pp_s, xr, pp_d, cpp, T_PP, E_PP, out + OFF_PPZ, 0);
    runHead(pl_count, hw_plc, xl, pl_s, xr, pl_d, cpl, T_PL, E_PL, out + OFF_PLC, 2);
    runHead(pl_zero,  hw_plz, xl, pl_s, xr, pl_d, cpl, T_PL, E_PL, out + OFF_PLZ, 0);
}
